// Attention_1855425872515
// MI455X (gfx1250) — matrix operand order repaired and checked by reading, not compiled, not run on hardware
//
#include <hip/hip_runtime.h>
#include <hip/hip_bf16.h>
#include <math.h>

// ---------------- problem constants ----------------
#define Bb   2
#define Ss   2048
#define Dd   1024
#define Hh   16
#define KHh  4
#define HDd  64
#define NQ   (Hh * HDd)    // 1024
#define NKV  (KHh * HDd)   // 256
#define MROWS (Bb * Ss)    // 4096

typedef __attribute__((ext_vector_type(16))) _Float16 v16h;
typedef __attribute__((ext_vector_type(8)))  _Float16 v8h;
typedef __attribute__((ext_vector_type(8)))  float    v8f;

// ---------------- WMMA helper ----------------
__device__ __forceinline__ v8f wmma_f16(v16h a, v16h b, v8f c) {
  return __builtin_amdgcn_wmma_f32_16x16x32_f16(
      false, a, false, b, (short)0, c, false, false);
}

// ---------------- CDNA5 async global->LDS copy (ASYNCcnt) ----------------
__device__ __forceinline__ void async_b128(unsigned lds_off, const void* g) {
  unsigned long long ga = (unsigned long long)(uintptr_t)g;
  asm volatile("global_load_async_to_lds_b128 %0, %1, off"
               :: "v"(lds_off), "v"(ga) : "memory");
}
__device__ __forceinline__ void wait_async0() {
  asm volatile("s_wait_asynccnt 0x0" ::: "memory");
}
__device__ __forceinline__ void wait_ds0() {
  asm volatile("s_wait_dscnt 0x0" ::: "memory");
}

// A fragment (16x32 f16) from f16 row-major [row][k], row stride ld.
__device__ __forceinline__ v16h a_frag_f16(const _Float16* p, int ld,
                                           int row_base, int k_base) {
  const int lane = threadIdx.x & 31;
  const int row  = row_base + (lane & 15);
  const int h    = lane >> 4;
  const _Float16* q0 = p + (size_t)row * ld + k_base + h * 8;
  v8h lo = *(const v8h*)q0;
  v8h hi = *(const v8h*)(q0 + 16);
  v16h a;
#pragma unroll
  for (int i = 0; i < 8; ++i) { a[i] = lo[i]; a[i + 8] = hi[i]; }
  return a;
}

// B fragment (32x16 f16): element B[k][n] = p[(col_base+n)*ld + k_base + k].
__device__ __forceinline__ v16h b_frag_f16(const _Float16* p, int ld,
                                           int col_base, int k_base) {
  const int lane = threadIdx.x & 31;
  const int n    = lane & 15;
  const int h    = lane >> 4;
  const _Float16* q0 = p + (size_t)(col_base + n) * ld + k_base + h * 8;
  v8h lo = *(const v8h*)q0;
  v8h hi = *(const v8h*)(q0 + 16);
  v16h b;
#pragma unroll
  for (int i = 0; i < 8; ++i) { b[i] = lo[i]; b[i + 8] = hi[i]; }
  return b;
}

// A fragment from a 16x32 f16 tile in LDS (row stride 32 halves).
__device__ __forceinline__ v16h a_frag_lds(const _Float16* p) {
  const int lane = threadIdx.x & 31;
  const int row  = lane & 15;
  const int h    = lane >> 4;
  v8h lo = *(const v8h*)(p + row * 32 + h * 8);
  v8h hi = *(const v8h*)(p + row * 32 + 16 + h * 8);
  v16h a;
#pragma unroll
  for (int i = 0; i < 8; ++i) { a[i] = lo[i]; a[i + 8] = hi[i]; }
  return a;
}

// ---------------- prep kernels ----------------
// f32 [K][N] -> f16 [N][K]
__global__ void wt_transpose_kernel(const float* __restrict__ in,
                                    _Float16* __restrict__ out,
                                    int K, int N) {
  int idx = blockIdx.x * blockDim.x + threadIdx.x;
  if (idx >= K * N) return;
  int k = idx / N;
  int n = idx - k * N;
  out[(size_t)n * K + k] = (_Float16)in[(size_t)k * N + n];
}

// elementwise f32 -> f16
__global__ void f32_to_f16_kernel(const float* __restrict__ in,
                                  _Float16* __restrict__ out, int nelem) {
  int i = blockIdx.x * blockDim.x + threadIdx.x;
  if (i < nelem) out[i] = (_Float16)in[i];
}

// ---------------- kernel 1: fused QKV projection + RoPE ----------------
// One wave computes a 32x64 output tile (8 WMMA per 32-deep K step).
// grid.x = 24 (64-col tiles: 0..15 q, 16..19 k, 20..23 v), grid.y = 128.
__global__ void qkv_rope_kernel(const _Float16* __restrict__ xh,
                                const _Float16* __restrict__ wq_t,
                                const _Float16* __restrict__ wk_t,
                                const _Float16* __restrict__ wv_t,
                                const float* __restrict__ fcos,
                                const float* __restrict__ fsin,
                                _Float16* __restrict__ qf,   // [MROWS][NQ]
                                _Float16* __restrict__ kf,   // [MROWS][NKV]
                                _Float16* __restrict__ vtf)  // [B][NKV][S]
{
  const int nt   = blockIdx.x;
  const int mt   = blockIdx.y;
  const int lane = threadIdx.x & 31;
  const int n    = lane & 15;
  const int half = lane >> 4;
  const int mbase = mt * 32;

  const _Float16* wt;
  int ncol, kind;                       // 0 = q, 1 = k, 2 = v
  if (nt < 16)      { wt = wq_t; ncol = nt * 64;        kind = 0; }
  else if (nt < 20) { wt = wk_t; ncol = (nt - 16) * 64; kind = 1; }
  else              { wt = wv_t; ncol = (nt - 20) * 64; kind = 2; }

  v8f acc[2][4] = {};
#pragma unroll 2
  for (int kk = 0; kk < Dd; kk += 32) {
    v16h a0 = a_frag_f16(xh, Dd, mbase, kk);
    v16h a1 = a_frag_f16(xh, Dd, mbase + 16, kk);
#pragma unroll
    for (int j = 0; j < 4; ++j) {
      v16h b = b_frag_f16(wt, Dd, ncol + j * 16, kk);
      acc[0][j] = wmma_f16(a0, b, acc[0][j]);
      acc[1][j] = wmma_f16(a1, b, acc[1][j]);
    }
  }

#pragma unroll
  for (int mi = 0; mi < 2; ++mi) {
#pragma unroll
    for (int j = 0; j < 4; ++j) {
      const int col = ncol + j * 16 + n;
      if (kind < 2) {
        // RoPE: interleaved pairs (2p,2p+1) sit in adjacent lanes.
        const int pr = (col & (HDd - 1)) >> 1;
#pragma unroll
        for (int r = 0; r < 8; ++r) {
          const int m = mbase + mi * 16 + r + 8 * half;
          const int s = m & (Ss - 1);
          const float c  = fcos[s * (HDd / 2) + pr];
          const float sn = fsin[s * (HDd / 2) + pr];
          const float self  = acc[mi][j][r];
          const float other = __shfl_xor(self, 1, 32);
          const float o = (col & 1) ? (other * sn + self * c)
                                    : (self * c - other * sn);
          if (kind == 0) qf[(size_t)m * NQ + col]  = (_Float16)o;
          else           kf[(size_t)m * NKV + col] = (_Float16)o;
        }
      } else {
        // V stored transposed [b][dim][s] so P*V B-fragments are contiguous.
#pragma unroll
        for (int r = 0; r < 8; ++r) {
          const int m  = mbase + mi * 16 + r + 8 * half;
          const int bb = m >> 11;
          const int s  = m & (Ss - 1);
          vtf[((size_t)bb * NKV + col) * Ss + s] = (_Float16)acc[mi][j][r];
        }
      }
    }
  }
}

// ---------------- kernel 2: flash attention (causal, GQA) ----------------
// One wave per (b, head, 16-query tile); 32 keys / iteration.
// K/V tiles staged into LDS with async copies, double-buffered.
__global__ void flash_attn_kernel(const _Float16* __restrict__ qf,
                                  const _Float16* __restrict__ kf,
                                  const _Float16* __restrict__ vtf,
                                  _Float16* __restrict__ attn) // [MROWS][NQ]
{
  __shared__ _Float16 kbuf[2][32 * 64];  // 2 x 4 KB  (key-major)
  __shared__ _Float16 vbuf[2][64 * 32];  // 2 x 4 KB  (dim-major)
  __shared__ _Float16 pbuf[16 * 32];     // 1 KB P bounce

  const int qtile = blockIdx.x;
  const int h     = blockIdx.y;
  const int b     = blockIdx.z;
  const int kvh   = h >> 2;              // REP = 4
  const int lane  = threadIdx.x & 31;
  const int n     = lane & 15;
  const int half  = lane >> 4;
  const int qbase = qtile * 16;

  const _Float16* qp = qf  + (size_t)b * Ss * NQ + h * HDd;
  const _Float16* kp = kf  + (size_t)b * Ss * NKV + kvh * HDd;
  const _Float16* vp = vtf + ((size_t)b * NKV + kvh * HDd) * Ss;

  // stage one 32-key K tile + V tile into LDS buffer `bf` (16 async b128s)
  auto stage = [&](int kt, int bf) {
    const int krow = lane >> 3;          // 0..3
    const int kdof = (lane & 7) * 8;     // dim offset (halves)
#pragma unroll
    for (int i = 0; i < 8; ++i) {
      const int rk = i * 4 + krow;       // relative key 0..31
      async_b128((unsigned)(uintptr_t)&kbuf[bf][rk * 64 + kdof],
                 kp + (size_t)(kt + rk) * NKV + kdof);
    }
    const int vrow = lane >> 2;          // 0..7
    const int vkof = (lane & 3) * 8;     // key offset (halves)
#pragma unroll
    for (int i = 0; i < 8; ++i) {
      const int rd = i * 8 + vrow;       // dim 0..63
      async_b128((unsigned)(uintptr_t)&vbuf[bf][rd * 32 + vkof],
                 vp + (size_t)rd * Ss + kt + vkof);
    }
  };

  // Q fragments: HD=64 -> two 32-deep A fragments, loaded once from global.
  const v16h qa0 = a_frag_f16(qp, NQ, qbase, 0);
  const v16h qa1 = a_frag_f16(qp, NQ, qbase, 32);

  float mrow[8], lrow[8];
#pragma unroll
  for (int r = 0; r < 8; ++r) { mrow[r] = -1e30f; lrow[r] = 0.0f; }
  v8f o0 = {}, o1 = {}, o2 = {}, o3 = {};

  const float scale = 0.125f;            // 1/sqrt(64)

  stage(0, 0);
  int buf = 0;
  const int kend = qbase + 16;
  for (int kt = 0; kt < kend; kt += 32) {
    wait_async0();                       // current buffer resident
    const _Float16* kb = &kbuf[buf][0];
    const _Float16* vb = &vbuf[buf][0];

    // ---- S = Q K^T for 32 keys (two 16x16 score tiles) ----
    v8f s0 = {}, s1 = {};
    s0 = wmma_f16(qa0, b_frag_f16(kb, 64, 0, 0),   s0);
    s0 = wmma_f16(qa1, b_frag_f16(kb, 64, 0, 32),  s0);
    s1 = wmma_f16(qa0, b_frag_f16(kb, 64, 16, 0),  s1);
    s1 = wmma_f16(qa1, b_frag_f16(kb, 64, 16, 32), s1);

    // ---- online softmax update ----
#pragma unroll
    for (int r = 0; r < 8; ++r) {
      const int qrow = qbase + r + 8 * half;
      float v0 = s0[r] * scale;
      float v1 = s1[r] * scale;
      if (kt + n > qrow)      v0 = -1e30f;     // causal mask
      if (kt + 16 + n > qrow) v1 = -1e30f;
      float rm = fmaxf(v0, v1);
#pragma unroll
      for (int off = 1; off < 16; off <<= 1)
        rm = fmaxf(rm, __shfl_xor(rm, off, 32));
      const float mnew = fmaxf(mrow[r], rm);
      const float p0 = __expf(v0 - mnew);
      const float p1 = __expf(v1 - mnew);
      const float corr = __expf(mrow[r] - mnew);
      float rs = p0 + p1;
#pragma unroll
      for (int off = 1; off < 16; off <<= 1)
        rs += __shfl_xor(rs, off, 32);
      lrow[r] = lrow[r] * corr + rs;
      mrow[r] = mnew;
      o0[r] *= corr; o1[r] *= corr; o2[r] *= corr; o3[r] *= corr;
      pbuf[(r + 8 * half) * 32 + n]      = (_Float16)p0;
      pbuf[(r + 8 * half) * 32 + 16 + n] = (_Float16)p1;
    }
    wait_ds0();   // P visible; also retires all prior DS reads (WAR-safe)

    // overlap: kick off next tile's async copies over the P*V math
    if (kt + 32 < kend) stage(kt + 32, buf ^ 1);

    // ---- O += P V  (A = P 16x32 from LDS, B = V strips from LDS) ----
    const v16h pa = a_frag_lds(pbuf);
    o0 = wmma_f16(pa, b_frag_f16(vb, 32, 0,  0), o0);
    o1 = wmma_f16(pa, b_frag_f16(vb, 32, 16, 0), o1);
    o2 = wmma_f16(pa, b_frag_f16(vb, 32, 32, 0), o2);
    o3 = wmma_f16(pa, b_frag_f16(vb, 32, 48, 0), o3);

    buf ^= 1;
  }

  // ---- normalize and store f16 attention output ----
  _Float16* op = attn + (size_t)b * Ss * NQ + h * HDd;
#pragma unroll
  for (int r = 0; r < 8; ++r) {
    const float inv = 1.0f / lrow[r];
    const size_t row = (size_t)(qbase + r + 8 * half) * NQ;
    op[row + n]      = (_Float16)(o0[r] * inv);
    op[row + 16 + n] = (_Float16)(o1[r] * inv);
    op[row + 32 + n] = (_Float16)(o2[r] * inv);
    op[row + 48 + n] = (_Float16)(o3[r] * inv);
  }
}

// ---------------- kernel 3: output projection ----------------
// One wave computes a 32x64 tile. grid.x = 64, grid.y = 128.
__global__ void out_proj_kernel(const _Float16* __restrict__ attn,
                                const _Float16* __restrict__ wo_t,
                                float* __restrict__ out) // [MROWS][D] f32
{
  const int nt   = blockIdx.x;
  const int mt   = blockIdx.y;
  const int lane = threadIdx.x & 31;
  const int n    = lane & 15;
  const int half = lane >> 4;
  const int mbase = mt * 32;
  const int ncol  = nt * 64;

  v8f acc[2][4] = {};
#pragma unroll 2
  for (int kk = 0; kk < NQ; kk += 32) {
    v16h a0 = a_frag_f16(attn, NQ, mbase, kk);
    v16h a1 = a_frag_f16(attn, NQ, mbase + 16, kk);
#pragma unroll
    for (int j = 0; j < 4; ++j) {
      v16h b = b_frag_f16(wo_t, NQ, ncol + j * 16, kk);
      acc[0][j] = wmma_f16(a0, b, acc[0][j]);
      acc[1][j] = wmma_f16(a1, b, acc[1][j]);
    }
  }
#pragma unroll
  for (int mi = 0; mi < 2; ++mi)
#pragma unroll
    for (int j = 0; j < 4; ++j)
#pragma unroll
      for (int r = 0; r < 8; ++r) {
        const size_t m = (size_t)(mbase + mi * 16 + r + 8 * half);
        out[m * Dd + ncol + j * 16 + n] = acc[mi][j][r];
      }
}

// ---------------- launcher ----------------
extern "C" void kernel_launch(void* const* d_in, const int* in_sizes, int n_in,
                              void* d_out, int out_size, void* d_ws, size_t ws_size,
                              hipStream_t stream) {
  const float* x    = (const float*)d_in[0];
  const float* wq   = (const float*)d_in[1];
  const float* wk   = (const float*)d_in[2];
  const float* wv   = (const float*)d_in[3];
  const float* wo   = (const float*)d_in[4];
  const float* fcos = (const float*)d_in[5];
  const float* fsin = (const float*)d_in[6];
  float* out = (float*)d_out;

  // ---- workspace carve (f16 buffers) ----
  char* ws = (char*)d_ws;
  size_t off = 0;
  _Float16* wq_t = (_Float16*)(ws + off); off += (size_t)NQ  * Dd * 2;
  _Float16* wk_t = (_Float16*)(ws + off); off += (size_t)NKV * Dd * 2;
  _Float16* wv_t = (_Float16*)(ws + off); off += (size_t)NKV * Dd * 2;
  _Float16* wo_t = (_Float16*)(ws + off); off += (size_t)Dd  * NQ * 2;
  _Float16* xh   = (_Float16*)(ws + off); off += (size_t)MROWS * Dd * 2;
  _Float16* qf   = (_Float16*)(ws + off); off += (size_t)MROWS * NQ * 2;
  _Float16* kf   = (_Float16*)(ws + off); off += (size_t)MROWS * NKV * 2;
  _Float16* vtf  = (_Float16*)(ws + off); off += (size_t)Bb * NKV * Ss * 2;
  _Float16* attn = (_Float16*)(ws + off); off += (size_t)MROWS * NQ * 2;

  // ---- 0) weight transpose->f16 and x->f16 ----
  {
    int e = Dd * NQ;
    wt_transpose_kernel<<<(e + 255) / 256, 256, 0, stream>>>(wq, wq_t, Dd, NQ);
    e = Dd * NKV;
    wt_transpose_kernel<<<(e + 255) / 256, 256, 0, stream>>>(wk, wk_t, Dd, NKV);
    wt_transpose_kernel<<<(e + 255) / 256, 256, 0, stream>>>(wv, wv_t, Dd, NKV);
    e = NQ * Dd;
    wt_transpose_kernel<<<(e + 255) / 256, 256, 0, stream>>>(wo, wo_t, NQ, Dd);
    e = MROWS * Dd;
    f32_to_f16_kernel<<<(e + 255) / 256, 256, 0, stream>>>(x, xh, e);
  }

  // ---- 1) fused QKV projection + RoPE (+ V transpose) ----
  qkv_rope_kernel<<<dim3(24, MROWS / 32), 32, 0, stream>>>(
      xh, wq_t, wk_t, wv_t, fcos, fsin, qf, kf, vtf);

  // ---- 2) flash attention (async double-buffered K/V staging) ----
  flash_attn_kernel<<<dim3(Ss / 16, Hh, Bb), 32, 0, stream>>>(qf, kf, vtf, attn);

  // ---- 3) output projection ----
  out_proj_kernel<<<dim3(Dd / 16, MROWS / 32), 32, 0, stream>>>(attn, wo_t, out);
}
